// AttentionBlock_55929064128512
// MI455X (gfx1250) — matrix operand order repaired and checked by reading, not compiled, not run on hardware
//
#include <hip/hip_runtime.h>
#include <math.h>
#include <stdint.h>

// ---------------------------------------------------------------------------
// AttentionBlock for MI455X (gfx1250): bf16 WMMA, flash softmax with DPP
// reductions, async-LDS (ASYNCcnt) double-buffered K/V staging.
// B=4, N=2048, DIM=512, HEADS=8, HEAD_SIZE=64, HIDDEN=512
// 1/sqrt(HEAD_SIZE) is folded into Wq at conversion time.
// ---------------------------------------------------------------------------

typedef __bf16 bf16_t;
typedef __attribute__((ext_vector_type(16))) __bf16 v16bf;
typedef __attribute__((ext_vector_type(8)))  float  v8f;

#define BB     4
#define NN     2048
#define DIMK   512
#define NH     8
#define HS     64
#define HID    512
#define LDQKV  1536        // fused [Q|K|V] row stride
#define NBLK   (NN / 32)   // 64 key blocks of 32

union Frag {
    v16bf        v;
    unsigned int u[8];
};

__device__ __forceinline__ v8f zero8() {
    v8f z;
#pragma unroll
    for (int i = 0; i < 8; ++i) z[i] = 0.0f;
    return z;
}

// A-operand (16x32 bf16) per-lane K offset of half-pair v (pairs contiguous):
// lanes 0-15: K = {0..7, 16..23};  lanes 16-31: +8
__device__ __forceinline__ int koffA(int v, int hi) {
    int base = (v < 4) ? (2 * v) : (2 * v + 8);
    return base + (hi ? 8 : 0);
}
// B-operand (32x16 bf16): same K order as A: lanes 0-15 K = {0..7, 16..23}; lanes 16-31: +8
__device__ __forceinline__ int koffB(int v, int hi) {
    return ((v < 4) ? (2 * v) : (2 * v + 8)) + (hi ? 8 : 0);
}

// DPP lane-exchange on the VALU (no LDS traffic); 16-lane butterfly pieces.
template <int CTRL>
__device__ __forceinline__ float dppmov(float x) {
    return __builtin_bit_cast(
        float, __builtin_amdgcn_mov_dpp(__builtin_bit_cast(int, x),
                                        CTRL, 0xF, 0xF, true));
}
__device__ __forceinline__ float redmax16(float x) {
    x = fmaxf(x, dppmov<0xB1>(x));   // quad_perm [1,0,3,2]  (xor 1)
    x = fmaxf(x, dppmov<0x4E>(x));   // quad_perm [2,3,0,1]  (xor 2)
    x = fmaxf(x, dppmov<0x141>(x));  // row_half_mirror      (combine quads in 8)
    x = fmaxf(x, dppmov<0x140>(x));  // row_mirror           (combine 8s in 16)
    return x;
}
__device__ __forceinline__ float redsum16(float x) {
    x += dppmov<0xB1>(x);
    x += dppmov<0x4E>(x);
    x += dppmov<0x141>(x);
    x += dppmov<0x140>(x);
    return x;
}

// ---------------------------------------------------------------------------
// Elementwise converts
// ---------------------------------------------------------------------------
__global__ void cvt_f32_bf16(const float* __restrict__ src, bf16_t* __restrict__ dst, int n) {
    int i = blockIdx.x * blockDim.x + threadIdx.x;
    if (i < n) dst[i] = (bf16_t)src[i];
}

// src [K][Nc] f32 row-major -> dst [Nc][K] bf16 (N-major weights), times scale
__global__ void transpose_cvt(const float* __restrict__ src, bf16_t* __restrict__ dst,
                              int K, int Nc, float scale) {
    int i = blockIdx.x * blockDim.x + threadIdx.x;
    if (i < K * Nc) {
        int k = i / Nc, n = i % Nc;
        dst[(size_t)n * K + k] = (bf16_t)(src[i] * scale);
    }
}

// V section of fused QKV  ->  vt[((b*NH+h)*HS+d)*NN + n]  (key-contiguous rows)
__global__ void transpose_v_kernel(const bf16_t* __restrict__ qkv, bf16_t* __restrict__ vt) {
    int i = blockIdx.x * blockDim.x + threadIdx.x;   // over B*NH*HS*NN
    if (i < BB * NH * HS * NN) {
        int n = i % NN; int t = i / NN;
        int d = t % HS; t /= HS;
        int h = t % NH; int b = t / NH;
        vt[i] = qkv[(size_t)(b * NN + n) * LDQKV + 2 * HID + h * HS + d];
    }
}

// ---------------------------------------------------------------------------
// Generic bf16 WMMA GEMM: C[M][Nc] = A[M][K] * Bt[Nc][K]^T
// One wave computes a 32x64 tile: 2 A-frags x 4 B-frags -> 8 wmma / K-step.
// Fragment chunks are 16B-contiguous so loads lower to global_load_b128.
// ---------------------------------------------------------------------------
template <bool OUT_F32>
__global__ __launch_bounds__(128) void wmma_gemm(
    const bf16_t* __restrict__ A, const bf16_t* __restrict__ Bt,
    void* __restrict__ Cp, int M, int Ncols, int K, int n_tiles) {

    const int lane = threadIdx.x & 31;
    const int wave = threadIdx.x >> 5;
    const int tile = blockIdx.x * 4 + wave;
    const int mt = tile / n_tiles, nt = tile % n_tiles;
    const int m0 = mt * 32, n0 = nt * 64;
    const int lan = lane & 15, hi = lane >> 4;

    v8f acc[2][4];
#pragma unroll
    for (int i = 0; i < 2; ++i)
#pragma unroll
        for (int j = 0; j < 4; ++j) acc[i][j] = zero8();

    const bf16_t* arow0 = A + (size_t)(m0 + lan) * K;
    const bf16_t* arow1 = arow0 + (size_t)16 * K;
    const bf16_t* brow[4];
#pragma unroll
    for (int j = 0; j < 4; ++j) brow[j] = Bt + (size_t)(n0 + j * 16 + lan) * K;

    for (int k0 = 0; k0 < K; k0 += 32) {
        Frag a0, a1, b[4];
#pragma unroll
        for (int v = 0; v < 8; ++v) {
            const int ka = k0 + koffA(v, hi);
            a0.u[v] = *(const unsigned int*)(arow0 + ka);
            a1.u[v] = *(const unsigned int*)(arow1 + ka);
            const int kb = k0 + koffB(v, hi);
#pragma unroll
            for (int j = 0; j < 4; ++j)
                b[j].u[v] = *(const unsigned int*)(brow[j] + kb);
        }
#pragma unroll
        for (int j = 0; j < 4; ++j) {
            acc[0][j] = __builtin_amdgcn_wmma_f32_16x16x32_bf16(
                false, a0.v, false, b[j].v, (short)0, acc[0][j], false, false);
            acc[1][j] = __builtin_amdgcn_wmma_f32_16x16x32_bf16(
                false, a1.v, false, b[j].v, (short)0, acc[1][j], false, false);
        }
    }

#pragma unroll
    for (int i = 0; i < 2; ++i)
#pragma unroll
        for (int j = 0; j < 4; ++j)
#pragma unroll
            for (int r = 0; r < 8; ++r) {
                const int row = m0 + i * 16 + hi * 8 + r;
                const int col = n0 + j * 16 + lan;
                const float val = acc[i][j][r];
                if (OUT_F32) ((float*)Cp)[(size_t)row * Ncols + col] = val;
                else         ((bf16_t*)Cp)[(size_t)row * Ncols + col] = (bf16_t)val;
            }
}

// ---------------------------------------------------------------------------
// Flash attention. One block = 4 waves = 64 query rows, all same (b,h).
// K/V tiles (32 keys) are DMA'd into LDS with global_load_async_to_lds_b128,
// double-buffered (tile i+1 in flight while computing tile i).
// ---------------------------------------------------------------------------
__global__ __launch_bounds__(128) void flash_attn(
    const bf16_t* __restrict__ qkv, const bf16_t* __restrict__ vt,
    bf16_t* __restrict__ O) {

    __shared__ __align__(16) __bf16 shK[2][32 * 64];   // [key][dim]   4KB each
    __shared__ __align__(16) __bf16 shV[2][64 * 32];   // [dim][key]   4KB each
    __shared__ __align__(16) __bf16 ptile[4][16 * 32]; // per-wave P staging

    const int tid  = threadIdx.x;
    const int lane = tid & 31;
    const int wave = tid >> 5;
    const int qb = blockIdx.x % (NN / 64);
    const int bh = blockIdx.x / (NN / 64);
    const int h = bh % NH, b = bh / NH;
    const int lan = lane & 15, hi = lane >> 4;

    // ---- Q fragments for this wave's 16 query rows (loaded once;
    //      already scaled by 1/sqrt(64) via Wq) ----
    const int qrow = b * NN + qb * 64 + wave * 16 + lan;
    const bf16_t* qbase = qkv + (size_t)qrow * LDQKV + h * HS;
    Frag qf[2];
#pragma unroll
    for (int s = 0; s < 2; ++s)
#pragma unroll
        for (int v = 0; v < 8; ++v)
            qf[s].u[v] = *(const unsigned int*)(qbase + s * 32 + koffA(v, hi));

    // ---- async-copy byte-offset bases (32-bit offsets vs. saddr base) ----
    const uint32_t kgbase = (uint32_t)(((uint32_t)(b * NN) * LDQKV + HID + h * HS) * 2);
    const uint32_t vgbase = (uint32_t)(((uint32_t)((b * NH + h) * HS) * NN) * 2);

    // Each thread DMAs 2 x 16B of K and 2 x 16B of V per tile (256 chunks/tile).
    auto issue_tile = [&](int kb, int sel) {
#pragma unroll
        for (int i = 0; i < 2; ++i) {
            const int ch = tid + i * 128;            // 0..255
            {   // K tile: [32 keys][64 dims], 128B/key = 8 chunks
                const int key = ch >> 3, o = ch & 7;
                const uint32_t voff = kgbase + (uint32_t)(kb + key) * (LDQKV * 2) + o * 16;
                const uint32_t ldst = (uint32_t)(uintptr_t)(&shK[sel][0]) + ch * 16;
                asm volatile("global_load_async_to_lds_b128 %0, %1, %2"
                             :: "v"(ldst), "v"(voff), "s"(qkv) : "memory");
            }
            {   // V^T tile: [64 dims][32 keys], 64B/dim = 4 chunks
                const int d = ch >> 2, o = ch & 3;
                const uint32_t voff = vgbase + (uint32_t)d * (NN * 2) + (uint32_t)kb * 2 + o * 16;
                const uint32_t ldst = (uint32_t)(uintptr_t)(&shV[sel][0]) + ch * 16;
                asm volatile("global_load_async_to_lds_b128 %0, %1, %2"
                             :: "v"(ldst), "v"(voff), "s"(vt) : "memory");
            }
        }
    };

    issue_tile(0, 0);      // prologue: tiles 0 and 1 in flight
    issue_tile(32, 1);

    float rowmax[8], rowsum[8];
#pragma unroll
    for (int r = 0; r < 8; ++r) { rowmax[r] = -INFINITY; rowsum[r] = 0.0f; }
    v8f oacc[4];
#pragma unroll
    for (int jj = 0; jj < 4; ++jj) oacc[jj] = zero8();

    __bf16* lds = &ptile[wave][0];

    for (int ib = 0; ib < NBLK; ++ib) {
        const int sel = ib & 1;
        // Wait for tile ib's 4 async copies (tile ib+1's 4 may stay in flight).
        if (ib + 1 < NBLK) asm volatile("s_wait_asynccnt 0x4" ::: "memory");
        else               asm volatile("s_wait_asynccnt 0x0" ::: "memory");
        __syncthreads();

        // ---- batch-load all K fragments and prefetch all V fragments so
        //      WMMAs issue back-to-back and V latency hides under softmax ----
        Frag kf[2][2];
#pragma unroll
        for (int j = 0; j < 2; ++j) {
            const __bf16* krow = &shK[sel][(j * 16 + lan) * 64];
#pragma unroll
            for (int s = 0; s < 2; ++s)
#pragma unroll
                for (int v = 0; v < 8; ++v)
                    kf[j][s].u[v] = *(const unsigned int*)(krow + s * 32 + koffB(v, hi));
        }
        Frag vf[4];
#pragma unroll
        for (int jj = 0; jj < 4; ++jj) {
            const __bf16* vrow = &shV[sel][(jj * 16 + lan) * 32];
#pragma unroll
            for (int v = 0; v < 8; ++v)
                vf[jj].u[v] = *(const unsigned int*)(vrow + koffB(v, hi));
        }

        // ---- S = Qs K^T for this 32-key block (two 16x16 tiles) ----
        v8f sacc[2];
        sacc[0] = zero8(); sacc[1] = zero8();
#pragma unroll
        for (int j = 0; j < 2; ++j)
#pragma unroll
            for (int s = 0; s < 2; ++s)
                sacc[j] = __builtin_amdgcn_wmma_f32_16x16x32_bf16(
                    false, qf[s].v, false, kf[j][s].v, (short)0, sacc[j], false, false);

        // ---- online softmax; rows live in 16-lane groups -> DPP butterflies
        float scl[8], p0v[8], p1v[8];
#pragma unroll
        for (int r = 0; r < 8; ++r) {
            const float m  = redmax16(fmaxf(sacc[0][r], sacc[1][r]));
            const float nm = fmaxf(rowmax[r], m);
            scl[r] = __expf(rowmax[r] - nm);
            rowmax[r] = nm;
            const float e0 = __expf(sacc[0][r] - nm);
            const float e1 = __expf(sacc[1][r] - nm);
            p0v[r] = e0; p1v[r] = e1;
            rowsum[r] = rowsum[r] * scl[r] + redsum16(e0 + e1);
        }
#pragma unroll
        for (int jj = 0; jj < 4; ++jj)
#pragma unroll
            for (int r = 0; r < 8; ++r) oacc[jj][r] *= scl[r];

        // ---- stage P through LDS: C-layout -> A-fragment layout ----
#pragma unroll
        for (int r = 0; r < 8; ++r) {
            const int m = hi * 8 + r;
            lds[m * 32 + lan]      = (bf16_t)p0v[r];
            lds[m * 32 + 16 + lan] = (bf16_t)p1v[r];
        }
        asm volatile("s_wait_dscnt 0x0" ::: "memory");

        Frag pa;
#pragma unroll
        for (int v = 0; v < 8; ++v)
            pa.u[v] = *(const unsigned int*)(lds + lan * 32 + koffA(v, hi));

        // ---- O += P * V   (K=32 keys, N=64 dims -> 4 back-to-back wmma) ----
#pragma unroll
        for (int jj = 0; jj < 4; ++jj)
            oacc[jj] = __builtin_amdgcn_wmma_f32_16x16x32_bf16(
                false, pa.v, false, vf[jj].v, (short)0, oacc[jj], false, false);

        __syncthreads();   // everyone done reading buffer `sel`
        if (ib + 2 < NBLK) issue_tile((ib + 2) * 32, sel);  // refill it
    }

    // ---- normalize + store O (bf16) ----
#pragma unroll
    for (int r = 0; r < 8; ++r) {
        const float inv = 1.0f / rowsum[r];
        const int row = b * NN + qb * 64 + wave * 16 + hi * 8 + r;
#pragma unroll
        for (int jj = 0; jj < 4; ++jj)
            O[(size_t)row * HID + h * HS + jj * 16 + lan] =
                (bf16_t)(oacc[jj][r] * inv);
    }
}

// ---------------------------------------------------------------------------
extern "C" void kernel_launch(void* const* d_in, const int* in_sizes, int n_in,
                              void* d_out, int out_size, void* d_ws, size_t ws_size,
                              hipStream_t stream) {
    const float* x   = (const float*)d_in[0];
    const float* Wq  = (const float*)d_in[1];
    const float* Wkv = (const float*)d_in[2];
    const float* Wo  = (const float*)d_in[3];

    char* ws = (char*)d_ws;
    size_t off = 0;
    auto alloc = [&](size_t bytes) -> void* {
        void* p = ws + off;
        off += (bytes + 255) & ~(size_t)255;
        return p;
    };
    bf16_t* xb   = (bf16_t*)alloc((size_t)BB * NN * DIMK * 2);        // x in bf16
    bf16_t* wqkv = (bf16_t*)alloc((size_t)(3 * HID) * DIMK * 2);      // [Wq|Wkv]^T, N-major
    bf16_t* wot  = (bf16_t*)alloc((size_t)DIMK * HID * 2);            // Wo^T, N-major
    bf16_t* qkvb = (bf16_t*)alloc((size_t)BB * NN * LDQKV * 2);       // fused QKV
    bf16_t* vtb  = (bf16_t*)alloc((size_t)BB * NH * HS * NN * 2);     // V transposed per head
    bf16_t* ob   = (bf16_t*)alloc((size_t)BB * NN * HID * 2);         // attention output

    // 1) converts; 1/sqrt(HEAD_SIZE) folded into Wq
    const int nx = BB * NN * DIMK;
    cvt_f32_bf16<<<(nx + 255) / 256, 256, 0, stream>>>(x, xb, nx);
    transpose_cvt<<<(DIMK * HID + 255) / 256, 256, 0, stream>>>(Wq, wqkv, DIMK, HID, 0.125f);
    transpose_cvt<<<(DIMK * 2 * HID + 255) / 256, 256, 0, stream>>>(
        Wkv, wqkv + (size_t)HID * DIMK, DIMK, 2 * HID, 1.0f);
    transpose_cvt<<<(HID * DIMK + 255) / 256, 256, 0, stream>>>(Wo, wot, HID, DIMK, 1.0f);

    // 2) fused QKV projection: [8192 x 512] * [512 x 1536]
    {
        const int M = BB * NN, Nc = LDQKV, K = DIMK;
        const int ntiles = Nc / 64;                 // 24
        const int tiles = (M / 32) * ntiles;        // 6144
        wmma_gemm<false><<<tiles / 4, 128, 0, stream>>>(xb, wqkv, qkvb, M, Nc, K, ntiles);
    }

    // 3) V transpose for PV operand
    {
        const int nv = BB * NH * HS * NN;
        transpose_v_kernel<<<(nv + 255) / 256, 256, 0, stream>>>(qkvb, vtb);
    }

    // 4) flash attention (1024 blocks x 128 threads)
    {
        const int blocks = BB * NH * (NN / 64);
        flash_attn<<<blocks, 128, 0, stream>>>(qkvb, vtb, ob);
    }

    // 5) output projection: [8192 x 512] * [512 x 512] -> f32 d_out
    {
        const int M = BB * NN, Nc = DIMK, K = HID;
        const int ntiles = Nc / 64;                 // 8
        const int tiles = (M / 32) * ntiles;        // 2048
        wmma_gemm<true><<<tiles / 4, 128, 0, stream>>>(ob, wot, d_out, M, Nc, K, ntiles);
    }
}
